// Layer_14190571946521
// MI455X (gfx1250) — compile-verified, matrix operand order checked by reading, not run on hardware
//
#include <hip/hip_runtime.h>
#include <hip/hip_bf16.h>

typedef __attribute__((ext_vector_type(16))) _Float16 v16h;
typedef __attribute__((ext_vector_type(8)))  _Float16 v8h;
typedef __attribute__((ext_vector_type(8)))  float    v8f;

#define C_OLD 0.8944271909999159f
#define C_NEW 0.4472135954999579f

#define WMMA(a,b,c)  __builtin_amdgcn_wmma_f32_16x16x32_f16(false,(a),false,(b),(short)0,(c),false,false)

// ---------------------------------------------------------------------------
// Weight prep: fp32 [K][N] row-major -> f16 [N][Kpad] (K zero-padded)
// ---------------------------------------------------------------------------
__global__ void prep_transpose(const float* __restrict__ src, _Float16* __restrict__ dst,
                               int K, int N, int Kpad) {
  int idx = blockIdx.x * blockDim.x + threadIdx.x;
  if (idx >= N * Kpad) return;
  int n = idx / Kpad, k = idx - n * Kpad;
  dst[idx] = (k < K) ? (_Float16)src[k * N + n] : (_Float16)0.0f;
}
// N-padded variant: dst [Npad][Kpad], rows >= N are zero
__global__ void prep_transpose_pad(const float* __restrict__ src, _Float16* __restrict__ dst,
                                   int K, int N, int Kpad, int Npad) {
  int idx = blockIdx.x * blockDim.x + threadIdx.x;
  if (idx >= Npad * Kpad) return;
  int n = idx / Kpad, k = idx - n * Kpad;
  dst[idx] = (n < N && k < K) ? (_Float16)src[k * N + n] : (_Float16)0.0f;
}
__global__ void prep_copy(const float* __restrict__ src, _Float16* __restrict__ dst, int total) {
  int idx = blockIdx.x * blockDim.x + threadIdx.x;
  if (idx < total) dst[idx] = (_Float16)src[idx];
}

// ---------------------------------------------------------------------------
// Fragment helpers (wave32, 16x16x32 f16 WMMA layouts)
// ---------------------------------------------------------------------------
__device__ inline v16h load_frag(const _Float16* p) {
  v8h lo = *(const v8h*)p;
  v8h hi = *(const v8h*)(p + 16);
  return __builtin_shufflevector(lo, hi, 0,1,2,3,4,5,6,7,8,9,10,11,12,13,14,15);
}
__device__ inline v16h afrag(const _Float16* t, int ldk, int kc, int lane) {
  int row = lane & 15, hsel = (lane >> 4) & 1;
  return load_frag(t + row * ldk + kc * 32 + hsel * 8);
}
__device__ inline v16h bfrag(const _Float16* w, int ldk, int nt, int kc, int lane) {
  int col = nt * 16 + (lane & 15), hsel = (lane >> 4) & 1;
  return load_frag(w + col * ldk + kc * 32 + hsel * 8);
}
__device__ inline void cstore(float* t, int ldn, int nt, v8f c, int lane) {
  int col = nt * 16 + (lane & 15);
  int mb  = ((lane >> 4) & 1) ? 8 : 0;
#pragma unroll
  for (int r = 0; r < 8; ++r) t[(mb + r) * ldn + col] = c[r];
}

// ---------------------------------------------------------------------------
struct Params {
  const float *latents, *node_features, *edge_features, *cutoff, *one_hot;
  const float *wig1, *wig2, *mole;
  const float *b0, *be, *ln_g, *ln_b;
  const float *L1b1, *L1b2, *L1b3, *L2b1, *L2b2, *L2b3;
  const int   *edge_index, *active_edges;
  const _Float16 *WRT;
  const _Float16 *W0T, *W1RT, *W1IT, *W2RT, *W2IT;
  const _Float16 *P0T, *P1T, *P2T, *WET;
  const _Float16 *L11, *L12, *L13, *L21, *L22, *L23;
  const _Float16 *T0H, *T1H, *T2H;
  float* out;
  int E;
};

// LDS layout (bytes)
#define OFF_GATE  0       // 16*12 f32
#define OFF_LATF  768     // 16*128 f32
#define OFF_LATH  8960    // 16*128 f16  (later: lat_ln f16)
#define OFF_OH    13056   // 16*128 f16
#define OFF_SC32  17152   // 16*16 f32 (scalars = p0 pre-scale)
#define OFF_CUT   18176   // 16 f32
#define OFF_MU    18240   // 32 f32 (mu, rstd)
#define OFF_IDX   18368   // 48 int (ae, sidx, didx)
#define OFF_ARENA 18560
#define ARENA_SZ  26624
#define SMEM_BYTES (OFF_ARENA + ARENA_SZ)   // 45184

__global__ __launch_bounds__(32)
void fused_edge_kernel(Params p) {
  __shared__ unsigned char smem[SMEM_BYTES] __attribute__((aligned(16)));
  const int lane = threadIdx.x;
  const int e0   = blockIdx.x * 16;

  float*     GATE = (float*)(smem + OFF_GATE);
  float*     LATF = (float*)(smem + OFF_LATF);
  _Float16*  LATH = (_Float16*)(smem + OFF_LATH);
  _Float16*  OH   = (_Float16*)(smem + OFF_OH);
  float*     SC32 = (float*)(smem + OFF_SC32);
  float*     CUT  = (float*)(smem + OFF_CUT);
  float*     MU   = (float*)(smem + OFF_MU);
  int*       AE   = (int*)(smem + OFF_IDX);
  int*       SIDX = AE + 16;
  int*       DIDX = AE + 32;
  unsigned char* A = smem + OFF_ARENA;

  // phase-A arena
  _Float16* IN0 = (_Float16*)(A + 0);       // 16x160
  _Float16* XP1 = (_Float16*)(A + 5120);    // 16x96
  _Float16* XM1 = (_Float16*)(A + 8192);    // 16x96
  _Float16* XP2 = (_Float16*)(A + 11264);   // 16x64
  _Float16* XM2 = (_Float16*)(A + 13312);   // 16x64
  float*    Y0  = (float*)(A + 15360);      // 16x80 (also gate-logit scratch in phase 0)
  float*    YP1 = (float*)(A + 20480);      // 16x32
  float*    YM1 = (float*)(A + 22528);      // 16x32
  float*    YP2 = (float*)(A + 24576);      // 16x16
  float*    YM2 = (float*)(A + 25600);      // 16x16
  // phase-B aliases
  _Float16* V1[3] = {(_Float16*)(A+0), (_Float16*)(A+1024), (_Float16*)(A+2048)};
  _Float16* V2[5] = {(_Float16*)(A+3072), (_Float16*)(A+4096), (_Float16*)(A+5120),
                     (_Float16*)(A+6144), (_Float16*)(A+7168)};
  _Float16* SCAL16 = (_Float16*)(A + 8192); // 16x32
  float*    S0T = (float*)(A + 9216);       // 16x16
  float*    S1T = (float*)(A + 10240);
  float*    S2T = (float*)(A + 11264);
  float*    WT  = (float*)(A + 12288);      // 16x48
  float*    P1L = (float*)(A + 15360);      // 16x48
  float*    P2L = (float*)(A + 18432);      // 16x80
  // MLP aliases
  _Float16* HA   = (_Float16*)(A + 0);      // 16x128
  _Float16* HB   = (_Float16*)(A + 4096);   // 16x128
  _Float16* SC16 = (_Float16*)(A + 8192);   // 16x32

  // ---------------- Phase 0: gathers + staging ----------------
  if (lane < 16) {
    int er = e0 + lane; if (er >= p.E) er = p.E - 1;
    int ae = p.active_edges[er];
    AE[lane]   = ae;
    SIDX[lane] = p.edge_index[ae];
    DIDX[lane] = p.edge_index[p.E + ae];
    CUT[lane]  = p.cutoff[ae];
  }
  __syncthreads();

#pragma unroll 4
  for (int i = 0; i < 64; ++i) {           // latents: fp32 + f16 copies
    int idx = i * 32 + lane, row = idx >> 7, col = idx & 127;
    float v = p.latents[(size_t)AE[row] * 128 + col];
    LATF[row * 128 + col] = v;
    LATH[row * 128 + col] = (_Float16)v;
  }
#pragma unroll 4
  for (int i = 0; i < 64; ++i) {           // edge_one_hot f16
    int idx = i * 32 + lane, row = idx >> 7, col = idx & 127;
    int er = e0 + row; if (er >= p.E) er = p.E - 1;
    OH[row * 128 + col] = (_Float16)p.one_hot[(size_t)er * 128 + col];
  }
  __syncthreads();

  // gate logits via WMMA: lat(16x128) @ WrT(128x16, cols 8..15 zero)
  {
    v8f C = {};
#pragma unroll
    for (int kc = 0; kc < 4; ++kc)
      C = WMMA(afrag(LATH, 128, kc, lane), bfrag(p.WRT, 128, 0, kc, lane), C);
    cstore(Y0, 16, 0, C, lane);            // Y0 region reused as logit scratch [16][16]
  }
  __syncthreads();
  if (lane < 16) {                          // softmax + skip "expert"
    float lg[8];
    float mx = -1e30f;
    for (int e = 0; e < 8; ++e) { lg[e] = Y0[lane * 16 + e] + p.mole[e]; mx = fmaxf(mx, lg[e]); }
    float s = 0.f;
    for (int e = 0; e < 8; ++e) { float v = __expf(lg[e] - mx); lg[e] = v; s += v; }
    float inv = 1.f / s;
    for (int e = 0; e < 8; ++e) GATE[lane * 12 + e] = lg[e] * inv;
    GATE[lane * 12 + 8] = 1.f; GATE[lane*12+9] = 0.f; GATE[lane*12+10] = 0.f; GATE[lane*12+11] = 0.f;
  }
  // zero K-pad columns of activation tiles
  for (int i = lane; i < 256; i += 32) {
    int r = i >> 4, c = i & 15;
    IN0[r * 160 + 144 + c] = (_Float16)0.f;
    XP2[r * 64 + 48 + c]   = (_Float16)0.f;
    XM2[r * 64 + 48 + c]   = (_Float16)0.f;
  }
  __syncthreads();

  // Wigner rotations + activation assembly (2 lanes per edge, 24 channels each)
  {
    int row = lane >> 1, half = lane & 1;
    int er = e0 + row; if (er >= p.E) er = p.E - 1;
    const float* srcp = p.node_features + (size_t)SIDX[row] * 144;
    const float* edgp = p.edge_features + (size_t)er * 144;
    const float* dstp = p.node_features + (size_t)DIDX[row] * 144;
    const float* w1 = p.wig1 + (size_t)er * 9;
    const float* w2 = p.wig2 + (size_t)er * 25;
    for (int cc = 0; cc < 24; ++cc) {
      int c = half * 24 + cc;
      int seg = c >> 4, m = c & 15;
      const float* base = (seg == 0) ? srcp : (seg == 1) ? edgp : dstp;
      float f0 = base[m];
      float f1[3], f2[5], x1[3], x2[5];
#pragma unroll
      for (int j = 0; j < 3; ++j) f1[j] = base[16 + m * 3 + j];
#pragma unroll
      for (int j = 0; j < 5; ++j) f2[j] = base[64 + m * 5 + j];
#pragma unroll
      for (int i = 0; i < 3; ++i) { float a = 0.f; for (int j = 0; j < 3; ++j) a += w1[i*3+j]*f1[j]; x1[i] = a; }
#pragma unroll
      for (int i = 0; i < 5; ++i) { float a = 0.f; for (int j = 0; j < 5; ++j) a += w2[i*5+j]*f2[j]; x2[i] = a; }
      IN0[row*160 + c]      = (_Float16)f0;
      IN0[row*160 + 48 + c] = (_Float16)x1[1];
      IN0[row*160 + 96 + c] = (_Float16)x2[2];
      XP1[row*96 + c]       = (_Float16)x1[2];
      XP1[row*96 + 48 + c]  = (_Float16)x2[3];
      XM1[row*96 + c]       = (_Float16)x1[0];
      XM1[row*96 + 48 + c]  = (_Float16)x2[1];
      XP2[row*64 + c]       = (_Float16)x2[4];
      XM2[row*64 + c]       = (_Float16)x2[0];
    }
  }
  __syncthreads();

  // ---------------- MoE0: in0(144->80), 9 "experts" -------------
  {
    v8f C0[5] = {};
    for (int e = 0; e < 9; ++e) {
      const _Float16* W = p.W0T + (size_t)e * 80 * 160;
      _Float16 hg = (_Float16)GATE[(lane & 15) * 12 + e];
#pragma unroll
      for (int kc = 0; kc < 5; ++kc) {
        v16h a = afrag(IN0, 160, kc, lane) * hg;
#pragma unroll
        for (int nt = 0; nt < 5; ++nt)
          C0[nt] = WMMA(a, bfrag(W, 160, nt, kc, lane), C0[nt]);
      }
    }
#pragma unroll
    for (int nt = 0; nt < 5; ++nt) cstore(Y0, 80, nt, C0[nt], lane);
  }
  // ---------------- MoE1: 96->32 (r/i combos) -------------------
  // yp1 = gate*(xp1@W1r) - gate*(xm1@W1i) ; ym1 = gate*(xp1@W1i) + gate*(xm1@W1r)
  // f16 WMMA has no A-negate modifier -> fold the minus into the gate scale.
  {
    v8f Cp[2] = {}, Cm[2] = {};
    for (int e = 0; e < 9; ++e) {
      const _Float16* Wr_ = p.W1RT + (size_t)e * 32 * 96;
      const _Float16* Wi_ = p.W1IT + (size_t)e * 32 * 96;
      _Float16 hg = (_Float16)GATE[(lane & 15) * 12 + e];
      _Float16 hgn = (_Float16)0.f - hg;
#pragma unroll
      for (int kc = 0; kc < 3; ++kc) {
        v16h apr = afrag(XP1, 96, kc, lane);
        v16h amr = afrag(XM1, 96, kc, lane);
        v16h ap  = apr * hg;
        v16h am  = amr * hg;
        v16h amn = amr * hgn;
#pragma unroll
        for (int nt = 0; nt < 2; ++nt) {
          v16h br = bfrag(Wr_, 96, nt, kc, lane);
          v16h bi = bfrag(Wi_, 96, nt, kc, lane);
          Cp[nt] = WMMA(ap,  br, Cp[nt]);
          Cp[nt] = WMMA(amn, bi, Cp[nt]);
          Cm[nt] = WMMA(ap,  bi, Cm[nt]);
          Cm[nt] = WMMA(am,  br, Cm[nt]);
        }
      }
    }
    cstore(YP1, 32, 0, Cp[0], lane); cstore(YP1, 32, 1, Cp[1], lane);
    cstore(YM1, 32, 0, Cm[0], lane); cstore(YM1, 32, 1, Cm[1], lane);
  }
  // ---------------- MoE2: 48->16 --------------------------------
  {
    v8f Cp = {}, Cm = {};
    for (int e = 0; e < 9; ++e) {
      const _Float16* Wr_ = p.W2RT + (size_t)e * 16 * 64;
      const _Float16* Wi_ = p.W2IT + (size_t)e * 16 * 64;
      _Float16 hg = (_Float16)GATE[(lane & 15) * 12 + e];
      _Float16 hgn = (_Float16)0.f - hg;
#pragma unroll
      for (int kc = 0; kc < 2; ++kc) {
        v16h apr = afrag(XP2, 64, kc, lane);
        v16h amr = afrag(XM2, 64, kc, lane);
        v16h ap  = apr * hg;
        v16h am  = amr * hg;
        v16h amn = amr * hgn;
        v16h br = bfrag(Wr_, 64, 0, kc, lane);
        v16h bi = bfrag(Wi_, 64, 0, kc, lane);
        Cp = WMMA(ap,  br, Cp);
        Cp = WMMA(amn, bi, Cp);
        Cm = WMMA(ap,  bi, Cm);
        Cm = WMMA(am,  br, Cm);
      }
    }
    cstore(YP2, 16, 0, Cp, lane); cstore(YM2, 16, 0, Cm, lane);
  }
  __syncthreads();

  // ---------------- back-rotation, gating, v tiles --------------
  for (int i = lane; i < 9 * 16 * 32; i += 32) ((_Float16*)(A))[i] = (_Float16)0.f;
  __syncthreads();
  {
    int row = lane >> 1, half = lane & 1;
    int er = e0 + row; if (er >= p.E) er = p.E - 1;
    const float* w1 = p.wig1 + (size_t)er * 9;
    const float* w2 = p.wig2 + (size_t)er * 25;
    for (int cc = 0; cc < 8; ++cc) {
      int ch = half * 8 + cc;
      float o1in[3] = { YM1[row*32+ch], Y0[row*80+48+ch], YP1[row*32+ch] };
      float o2in[5] = { YM2[row*16+ch], YM1[row*32+16+ch], Y0[row*80+64+ch],
                        YP1[row*32+16+ch], YP2[row*16+ch] };
      float g1 = 1.f / (1.f + __expf(-Y0[row*80+16+ch]));
      float g2 = 1.f / (1.f + __expf(-Y0[row*80+32+ch]));
      float sc = Y0[row*80+ch]; sc = sc / (1.f + __expf(-sc));
#pragma unroll
      for (int i = 0; i < 3; ++i) {
        float o = 0.f; for (int j = 0; j < 3; ++j) o += w1[j*3+i] * o1in[j];
        V1[i][row*32+ch] = (_Float16)(o * g1);
      }
#pragma unroll
      for (int i = 0; i < 5; ++i) {
        float o = 0.f; for (int j = 0; j < 5; ++j) o += w2[j*5+i] * o2in[j];
        V2[i][row*32+ch] = (_Float16)(o * g2);
      }
      SCAL16[row*32+ch] = (_Float16)sc;
    }
  }
  __syncthreads();

  // ---------------- channel mixes P0/P1/P2 ----------------------
  {
    v8f C = {};
    C = WMMA(afrag(SCAL16, 32, 0, lane), bfrag(p.P0T, 32, 0, 0, lane), C);
    int col = lane & 15, mb = ((lane >> 4) & 1) ? 8 : 0;
    float b = p.b0[col];
#pragma unroll
    for (int r = 0; r < 8; ++r) SC32[(mb + r) * 16 + col] = C[r] + b;
  }
#pragma unroll
  for (int m = 0; m < 3; ++m) {
    v8f C = {};
    C = WMMA(afrag(V1[m], 32, 0, lane), bfrag(p.P1T, 32, 0, 0, lane), C);
    int col = lane & 15, mb = ((lane >> 4) & 1) ? 8 : 0;
#pragma unroll
    for (int r = 0; r < 8; ++r) P1L[(mb + r) * 48 + col * 3 + m] = C[r];
  }
#pragma unroll
  for (int m = 0; m < 5; ++m) {
    v8f C = {};
    C = WMMA(afrag(V2[m], 32, 0, lane), bfrag(p.P2T, 32, 0, 0, lane), C);
    int col = lane & 15, mb = ((lane >> 4) & 1) ? 8 : 0;
#pragma unroll
    for (int r = 0; r < 8; ++r) P2L[(mb + r) * 80 + col * 5 + m] = C[r];
  }
  // w = lat @ We + be
  {
    v8f Cw[3] = {};
#pragma unroll
    for (int kc = 0; kc < 4; ++kc) {
      v16h a = afrag(LATH, 128, kc, lane);
#pragma unroll
      for (int nt = 0; nt < 3; ++nt)
        Cw[nt] = WMMA(a, bfrag(p.WET, 128, nt, kc, lane), Cw[nt]);
    }
#pragma unroll
    for (int nt = 0; nt < 3; ++nt) {
      int col = nt * 16 + (lane & 15), mb = ((lane >> 4) & 1) ? 8 : 0;
      float b = p.be[col];
#pragma unroll
      for (int r = 0; r < 8; ++r) WT[(mb + r) * 48 + col] = Cw[nt][r] + b;
    }
  }
  // tune scales s0/s1/s2 = one_hot @ T*.T
  {
    v8f Cs0 = {}, Cs1 = {}, Cs2 = {};
#pragma unroll
    for (int kc = 0; kc < 4; ++kc) {
      v16h a = afrag(OH, 128, kc, lane);
      Cs0 = WMMA(a, bfrag(p.T0H, 128, 0, kc, lane), Cs0);
      Cs1 = WMMA(a, bfrag(p.T1H, 128, 0, kc, lane), Cs1);
      Cs2 = WMMA(a, bfrag(p.T2H, 128, 0, kc, lane), Cs2);
    }
    cstore(S0T, 16, 0, Cs0, lane);
    cstore(S1T, 16, 0, Cs1, lane);
    cstore(S2T, 16, 0, Cs2, lane);
  }
  __syncthreads();

  // ---------------- edge output assembly ------------------------
  {
    int row = lane >> 1, base = (lane & 1) * 72;
    int er = e0 + row; if (er >= p.E) er = p.E - 1;
    const float* ef = p.edge_features + (size_t)er * 144;
    float* orow = p.out + (size_t)er * 272;
    for (int i = 0; i < 72; ++i) {
      int d = base + i;
      float pv, s;
      if (d < 16)      { pv = SC32[row*16 + d] * WT[row*48 + d];                   s = S0T[row*16 + d]; }
      else if (d < 64) { int q = d - 16, c = q / 3; pv = P1L[row*48+q]*WT[row*48+16+c]; s = S1T[row*16+c]; }
      else             { int q = d - 64, c = q / 5; pv = P2L[row*80+q]*WT[row*48+32+c]; s = S2T[row*16+c]; }
      float eo = C_OLD * ef[d] + C_NEW * pv;
      orow[d] = eo * (1.f + s);
    }
  }
  __syncthreads();

  // ---------------- latent branch: LN + two MLPs ----------------
  for (int i = lane; i < 16 * 32; i += 32) SC16[i] = (_Float16)0.f;
  __syncthreads();
  for (int i = lane; i < 16 * 16; i += 32) {
    int r = i >> 4, c = i & 15;
    SC16[r * 32 + c] = (_Float16)SC32[i];
  }
  if (lane < 16) {
    const float* lr = LATF + lane * 128;
    float s = 0.f, s2 = 0.f;
    for (int k = 0; k < 128; ++k) { float v = lr[k]; s += v; s2 += v * v; }
    float mu = s * (1.f / 128.f);
    float var = s2 * (1.f / 128.f) - mu * mu;
    MU[lane] = mu; MU[16 + lane] = rsqrtf(var + 1e-5f);
  }
  __syncthreads();
#pragma unroll 4
  for (int i = 0; i < 64; ++i) {           // lat_ln (overwrites LATH)
    int idx = i * 32 + lane, row = idx >> 7, col = idx & 127;
    float v = (LATF[row*128+col] - MU[row]) * MU[16+row] * p.ln_g[col] + p.ln_b[col];
    LATH[row * 128 + col] = (_Float16)v;
  }
  __syncthreads();

  v8f H[8];
  // L1 layer1: [lat_ln(128) | scalars(16 pad)] -> 128, silu
#pragma unroll
  for (int nt = 0; nt < 8; ++nt) H[nt] = (v8f){};
#pragma unroll
  for (int kc = 0; kc < 5; ++kc) {
    v16h a = (kc < 4) ? afrag(LATH, 128, kc, lane) : afrag(SC16, 32, 0, lane);
#pragma unroll
    for (int nt = 0; nt < 8; ++nt) H[nt] = WMMA(a, bfrag(p.L11, 160, nt, kc, lane), H[nt]);
  }
  {
    int col0 = lane & 15, mb = ((lane >> 4) & 1) ? 8 : 0;
#pragma unroll
    for (int nt = 0; nt < 8; ++nt) {
      int col = nt * 16 + col0; float b = p.L1b1[col];
#pragma unroll
      for (int r = 0; r < 8; ++r) { float v = H[nt][r] + b; v = v / (1.f + __expf(-v)); HA[(mb+r)*128+col] = (_Float16)v; }
    }
  }
  __syncthreads();
  // L1 layer2 -> HB silu
#pragma unroll
  for (int nt = 0; nt < 8; ++nt) H[nt] = (v8f){};
#pragma unroll
  for (int kc = 0; kc < 4; ++kc) {
    v16h a = afrag(HA, 128, kc, lane);
#pragma unroll
    for (int nt = 0; nt < 8; ++nt) H[nt] = WMMA(a, bfrag(p.L12, 128, nt, kc, lane), H[nt]);
  }
  {
    int col0 = lane & 15, mb = ((lane >> 4) & 1) ? 8 : 0;
#pragma unroll
    for (int nt = 0; nt < 8; ++nt) {
      int col = nt * 16 + col0; float b = p.L1b2[col];
#pragma unroll
      for (int r = 0; r < 8; ++r) { float v = H[nt][r] + b; v = v / (1.f + __expf(-v)); HB[(mb+r)*128+col] = (_Float16)v; }
    }
  }
  __syncthreads();
  // L1 layer3 -> HA (no act)
#pragma unroll
  for (int nt = 0; nt < 8; ++nt) H[nt] = (v8f){};
#pragma unroll
  for (int kc = 0; kc < 4; ++kc) {
    v16h a = afrag(HB, 128, kc, lane);
#pragma unroll
    for (int nt = 0; nt < 8; ++nt) H[nt] = WMMA(a, bfrag(p.L13, 128, nt, kc, lane), H[nt]);
  }
  {
    int col0 = lane & 15, mb = ((lane >> 4) & 1) ? 8 : 0;
#pragma unroll
    for (int nt = 0; nt < 8; ++nt) {
      int col = nt * 16 + col0; float b = p.L1b3[col];
#pragma unroll
      for (int r = 0; r < 8; ++r) HA[(mb+r)*128+col] = (_Float16)(H[nt][r] + b);
    }
  }
  __syncthreads();
  // L2 layer1: [h(128) | one_hot(128)] -> 128, silu -> HB
#pragma unroll
  for (int nt = 0; nt < 8; ++nt) H[nt] = (v8f){};
#pragma unroll
  for (int kc = 0; kc < 8; ++kc) {
    v16h a = (kc < 4) ? afrag(HA, 128, kc, lane) : afrag(OH, 128, kc - 4, lane);
#pragma unroll
    for (int nt = 0; nt < 8; ++nt) H[nt] = WMMA(a, bfrag(p.L21, 256, nt, kc, lane), H[nt]);
  }
  {
    int col0 = lane & 15, mb = ((lane >> 4) & 1) ? 8 : 0;
#pragma unroll
    for (int nt = 0; nt < 8; ++nt) {
      int col = nt * 16 + col0; float b = p.L2b1[col];
#pragma unroll
      for (int r = 0; r < 8; ++r) { float v = H[nt][r] + b; v = v / (1.f + __expf(-v)); HB[(mb+r)*128+col] = (_Float16)v; }
    }
  }
  __syncthreads();
  // L2 layer2 -> HA silu
#pragma unroll
  for (int nt = 0; nt < 8; ++nt) H[nt] = (v8f){};
#pragma unroll
  for (int kc = 0; kc < 4; ++kc) {
    v16h a = afrag(HB, 128, kc, lane);
#pragma unroll
    for (int nt = 0; nt < 8; ++nt) H[nt] = WMMA(a, bfrag(p.L22, 128, nt, kc, lane), H[nt]);
  }
  {
    int col0 = lane & 15, mb = ((lane >> 4) & 1) ? 8 : 0;
#pragma unroll
    for (int nt = 0; nt < 8; ++nt) {
      int col = nt * 16 + col0; float b = p.L2b2[col];
#pragma unroll
      for (int r = 0; r < 8; ++r) { float v = H[nt][r] + b; v = v / (1.f + __expf(-v)); HA[(mb+r)*128+col] = (_Float16)v; }
    }
  }
  __syncthreads();
  // L2 layer3 -> new_lat; combine + store
#pragma unroll
  for (int nt = 0; nt < 8; ++nt) H[nt] = (v8f){};
#pragma unroll
  for (int kc = 0; kc < 4; ++kc) {
    v16h a = afrag(HA, 128, kc, lane);
#pragma unroll
    for (int nt = 0; nt < 8; ++nt) H[nt] = WMMA(a, bfrag(p.L23, 128, nt, kc, lane), H[nt]);
  }
  {
    int col0 = lane & 15, mb = ((lane >> 4) & 1) ? 8 : 0;
#pragma unroll
    for (int nt = 0; nt < 8; ++nt) {
      int col = nt * 16 + col0; float b = p.L2b3[col];
#pragma unroll
      for (int r = 0; r < 8; ++r) {
        int row = mb + r;
        int er = e0 + row; if (er >= p.E) er = p.E - 1;
        float nl = C_NEW * CUT[row] * (H[nt][r] + b) + C_OLD * LATF[row * 128 + col];
        p.out[(size_t)er * 272 + 144 + col] = nl;
      }
    }
  }
}

// ---------------------------------------------------------------------------
// Workspace offsets (halves)
// ---------------------------------------------------------------------------
#define O_W0T  0ull
#define O_W1RT (O_W0T  + 9ull*80*160)
#define O_W1IT (O_W1RT + 9ull*32*96)
#define O_W2RT (O_W1IT + 9ull*32*96)
#define O_W2IT (O_W2RT + 9ull*16*64)
#define O_P0T  (O_W2IT + 9ull*16*64)
#define O_P1T  (O_P0T + 512ull)
#define O_P2T  (O_P1T + 512ull)
#define O_WET  (O_P2T + 512ull)
#define O_L11  (O_WET + 48ull*128)
#define O_L12  (O_L11 + 128ull*160)
#define O_L13  (O_L12 + 128ull*128)
#define O_L21  (O_L13 + 128ull*128)
#define O_L22  (O_L21 + 128ull*256)
#define O_L23  (O_L22 + 128ull*128)
#define O_T0   (O_L23 + 128ull*128)
#define O_T1   (O_T0 + 2048ull)
#define O_T2   (O_T1 + 2048ull)
#define O_WRT  (O_T2 + 2048ull)            // 16 x 128 (rows 8..15 zero)

extern "C" void kernel_launch(void* const* d_in, const int* in_sizes, int n_in,
                              void* d_out, int out_size, void* d_ws, size_t ws_size,
                              hipStream_t stream) {
  const float* latents  = (const float*)d_in[0];
  const float* node_f   = (const float*)d_in[1];
  const float* edge_f   = (const float*)d_in[3];
  const float* cutoff   = (const float*)d_in[5];
  const float* one_hot  = (const float*)d_in[6];
  const float* wig1     = (const float*)d_in[7];
  const float* wig2     = (const float*)d_in[8];
  const float* mole     = (const float*)d_in[9];
  const float* Wr       = (const float*)d_in[10];
  const float* W0       = (const float*)d_in[11];
  const float* W1r      = (const float*)d_in[12];
  const float* W1i      = (const float*)d_in[13];
  const float* W2r      = (const float*)d_in[14];
  const float* W2i      = (const float*)d_in[15];
  const float* S0       = (const float*)d_in[16];
  const float* S1r      = (const float*)d_in[17];
  const float* S1i      = (const float*)d_in[18];
  const float* S2r      = (const float*)d_in[19];
  const float* S2i      = (const float*)d_in[20];
  const float* P0       = (const float*)d_in[21];
  const float* b0       = (const float*)d_in[22];
  const float* P1       = (const float*)d_in[23];
  const float* P2       = (const float*)d_in[24];
  const float* We       = (const float*)d_in[25];
  const float* be       = (const float*)d_in[26];
  const float* ln_g     = (const float*)d_in[27];
  const float* ln_b     = (const float*)d_in[28];
  const float* L1w1     = (const float*)d_in[29];
  const float* L1b1     = (const float*)d_in[30];
  const float* L1w2     = (const float*)d_in[31];
  const float* L1b2     = (const float*)d_in[32];
  const float* L1w3     = (const float*)d_in[33];
  const float* L1b3     = (const float*)d_in[34];
  const float* L2w1     = (const float*)d_in[35];
  const float* L2b1     = (const float*)d_in[36];
  const float* L2w2     = (const float*)d_in[37];
  const float* L2b2     = (const float*)d_in[38];
  const float* L2w3     = (const float*)d_in[39];
  const float* L2b3     = (const float*)d_in[40];
  const float* T0       = (const float*)d_in[41];
  const float* T1       = (const float*)d_in[42];
  const float* T2       = (const float*)d_in[43];
  const int* edge_index = (const int*)d_in[44];
  const int* active_e   = (const int*)d_in[45];
  int E = in_sizes[0] / 128;

  _Float16* ws = (_Float16*)d_ws;
  auto tr = [&](const float* s, size_t off, int K, int N, int Kp) {
    int tot = N * Kp;
    prep_transpose<<<(tot + 255) / 256, 256, 0, stream>>>(s, ws + off, K, N, Kp);
  };
  for (int e = 0; e < 8; ++e) tr(W0 + (size_t)e*144*80, O_W0T + (size_t)e*80*160, 144, 80, 160);
  tr(S0, O_W0T + 8ull*80*160, 144, 80, 160);
  for (int e = 0; e < 8; ++e) tr(W1r + (size_t)e*96*32, O_W1RT + (size_t)e*32*96, 96, 32, 96);
  tr(S1r, O_W1RT + 8ull*32*96, 96, 32, 96);
  for (int e = 0; e < 8; ++e) tr(W1i + (size_t)e*96*32, O_W1IT + (size_t)e*32*96, 96, 32, 96);
  tr(S1i, O_W1IT + 8ull*32*96, 96, 32, 96);
  for (int e = 0; e < 8; ++e) tr(W2r + (size_t)e*48*16, O_W2RT + (size_t)e*16*64, 48, 16, 64);
  tr(S2r, O_W2RT + 8ull*16*64, 48, 16, 64);
  for (int e = 0; e < 8; ++e) tr(W2i + (size_t)e*48*16, O_W2IT + (size_t)e*16*64, 48, 16, 64);
  tr(S2i, O_W2IT + 8ull*16*64, 48, 16, 64);
  tr(P0, O_P0T, 16, 16, 32);
  tr(P1, O_P1T, 16, 16, 32);
  tr(P2, O_P2T, 16, 16, 32);
  tr(We, O_WET, 128, 48, 128);
  tr(L1w1, O_L11, 144, 128, 160);
  tr(L1w2, O_L12, 128, 128, 128);
  tr(L1w3, O_L13, 128, 128, 128);
  tr(L2w1, O_L21, 256, 128, 256);
  tr(L2w2, O_L22, 128, 128, 128);
  tr(L2w3, O_L23, 128, 128, 128);
  prep_copy<<<(2048 + 255) / 256, 256, 0, stream>>>(T0, ws + O_T0, 2048);
  prep_copy<<<(2048 + 255) / 256, 256, 0, stream>>>(T1, ws + O_T1, 2048);
  prep_copy<<<(2048 + 255) / 256, 256, 0, stream>>>(T2, ws + O_T2, 2048);
  prep_transpose_pad<<<(16*128 + 255) / 256, 256, 0, stream>>>(Wr, ws + O_WRT, 128, 8, 128, 16);

  Params pr;
  pr.latents = latents; pr.node_features = node_f; pr.edge_features = edge_f;
  pr.cutoff = cutoff; pr.one_hot = one_hot; pr.wig1 = wig1; pr.wig2 = wig2;
  pr.mole = mole;
  pr.b0 = b0; pr.be = be; pr.ln_g = ln_g; pr.ln_b = ln_b;
  pr.L1b1 = L1b1; pr.L1b2 = L1b2; pr.L1b3 = L1b3;
  pr.L2b1 = L2b1; pr.L2b2 = L2b2; pr.L2b3 = L2b3;
  pr.edge_index = edge_index; pr.active_edges = active_e;
  pr.WRT = ws + O_WRT;
  pr.W0T = ws + O_W0T; pr.W1RT = ws + O_W1RT; pr.W1IT = ws + O_W1IT;
  pr.W2RT = ws + O_W2RT; pr.W2IT = ws + O_W2IT;
  pr.P0T = ws + O_P0T; pr.P1T = ws + O_P1T; pr.P2T = ws + O_P2T;
  pr.WET = ws + O_WET;
  pr.L11 = ws + O_L11; pr.L12 = ws + O_L12; pr.L13 = ws + O_L13;
  pr.L21 = ws + O_L21; pr.L22 = ws + O_L22; pr.L23 = ws + O_L23;
  pr.T0H = ws + O_T0; pr.T1H = ws + O_T1; pr.T2H = ws + O_T2;
  pr.out = (float*)d_out; pr.E = E;

  int blocks = (E + 15) / 16;
  fused_edge_kernel<<<blocks, 32, 0, stream>>>(pr);
}
